// FullyConnected1_36962488549806
// MI455X (gfx1250) — compile-verified, matrix operand order checked by reading, not run on hardware
//
#include <hip/hip_runtime.h>
#include <hip/hip_bf16.h>

#define NN   1024   // sequence length (n, m dims)
#define DD   64     // feature dim
#define CC   128    // 2*DD channels into the 1x1 conv
#define HH   64     // output channels
#define MB   128    // m-columns per block
#define CPAD 136    // bf16 row stride (128 + 8) -> 272B, bank-conflict-free b128 loads

typedef __bf16 v16bf __attribute__((ext_vector_type(16)));
typedef float  v8f   __attribute__((ext_vector_type(8)));

union FragU { v16bf v; uint4 q[2]; };

// Gather a 16-element bf16 WMMA fragment: two contiguous 8-elem (16B) runs at
// p[0..7] and p[16..23], matching the 16-bit A/B operand VGPR layout
// (K = e + (e>=8 ? 8 : 0), per-lane base handled by caller).
__device__ __forceinline__ v16bf load_frag(const __bf16* p) {
    FragU u;
    u.q[0] = *(const uint4*)(p);
    u.q[1] = *(const uint4*)(p + 16);
    return u.v;
}

// ---------------------------------------------------------------------------
// Kernel 1: transpose z0 (N,D)->(D,N) f32, convert W to bf16, zero reductions
// ---------------------------------------------------------------------------
__global__ __launch_bounds__(256) void prep_kernel(
    const float* __restrict__ z0, const float* __restrict__ W,
    float* __restrict__ z_t, __bf16* __restrict__ w_bf,
    float* __restrict__ gsum, float* __restrict__ gsq)
{
    int idx = blockIdx.x * 256 + threadIdx.x;
    if (idx < DD * NN) {
        int d = idx / NN, n = idx % NN;
        z_t[idx] = z0[n * DD + d];                 // z_t[d][n]
    }
    if (idx < HH * CC) w_bf[idx] = (__bf16)W[idx]; // row-major [h][c]
    if (idx < HH) { gsum[idx] = 0.0f; gsq[idx] = 0.0f; }
}

// ---------------------------------------------------------------------------
// Kernel 2: for fixed n, 128 m's: build z_cat tile in LDS (bf16), WMMA against
// W, store un-normalized y to d_out, accumulate per-h sum / sumsq.
// Bias b[h] is skipped: it cancels exactly under InstanceNorm.
// ---------------------------------------------------------------------------
__global__ __launch_bounds__(256) void pairwise_gemm_kernel(
    const float* __restrict__ z_t,     // [DD][NN] f32
    const __bf16* __restrict__ w_bf,   // [HH][CC] bf16
    float* __restrict__ y,             // [HH][NN][NN] f32 (un-normalized)
    float* __restrict__ gsum, float* __restrict__ gsq)
{
    __shared__ __bf16 Btile[MB * CPAD]; // [m][c] bf16, padded rows
    __shared__ float  z_n[DD];
    __shared__ float  lsum[HH];
    __shared__ float  lsq[HH];

    const int n     = blockIdx.x;
    const int m0blk = blockIdx.y * MB;
    const int t     = threadIdx.x;

    if (t < DD) z_n[t] = z_t[t * NN + n];
    if (t < HH) { lsum[t] = 0.0f; lsq[t] = 0.0f; }
    __syncthreads();

    // ---- build feature tile: Btile[m][d] = |z[d,n]-z[d,m]|, Btile[m][64+d] = z[d,n]*z[d,m]
    {
        const int mloc  = t & (MB - 1);
        const int dbase = (t >> 7) * 32;           // 256 threads, 2 per column
        const int m     = m0blk + mloc;
        __bf16* row = &Btile[mloc * CPAD];
        #pragma unroll 8
        for (int i = 0; i < 32; ++i) {
            const int d  = dbase + i;
            const float v  = z_t[d * NN + m];      // coalesced across threads
            const float zn = z_n[d];
            row[d]      = (__bf16)fabsf(zn - v);
            row[DD + d] = (__bf16)(zn * v);
        }
    }
    __syncthreads();

    // ---- wave work assignment: 4 h-tiles x 2 m-halves
    const int wave = t >> 5, lane = t & 31;
    const int h0   = (wave & 3) << 4;              // 0,16,32,48
    const int msub = (wave >> 2) << 6;             // 0 or 64
    const int lm   = lane & 15;
    const int kbase = (lane & 16) ? 8 : 0;         // per-lane K offset in operand layout
    const int hadd  = (lane & 16) ? 8 : 0;         // D-matrix M offset for lanes 16..31

    // A fragments (W rows), loaded once, reused for all 4 m-tiles
    v16bf a[4];
    {
        const __bf16* wrow = w_bf + (h0 + lm) * CC + kbase;
        #pragma unroll
        for (int kc = 0; kc < 4; ++kc) a[kc] = load_frag(wrow + kc * 32);
    }

    float sacc[8], sqacc[8];
    #pragma unroll
    for (int r = 0; r < 8; ++r) { sacc[r] = 0.0f; sqacc[r] = 0.0f; }

    #pragma unroll
    for (int mt = 0; mt < 4; ++mt) {
        const int mloc = msub + mt * 16 + lm;
        const __bf16* brow = &Btile[mloc * CPAD + kbase];
        v8f acc = {0.0f, 0.0f, 0.0f, 0.0f, 0.0f, 0.0f, 0.0f, 0.0f};
        #pragma unroll
        for (int kc = 0; kc < 4; ++kc) {
            v16bf bf = load_frag(brow + kc * 32);
            acc = __builtin_amdgcn_wmma_f32_16x16x32_bf16(
                false, a[kc], false, bf, (short)0, acc, false, false);
        }
        // D layout: VGPR r -> h = h0 + hadd + r ; column m = lane&15
        const size_t m = (size_t)(m0blk + mloc);
        float* outp = y + (((size_t)(h0 + hadd) * NN + n) << 10) + m;
        #pragma unroll
        for (int r = 0; r < 8; ++r) {
            const float v = acc[r];
            outp[(size_t)r << 20] = v;             // h stride = NN*NN
            sacc[r]  += v;
            sqacc[r] += v * v;
        }
    }

    // ---- per-h block reduction in LDS (ds_add_f32), then one global atomic per h
    #pragma unroll
    for (int r = 0; r < 8; ++r) {
        const int h = h0 + hadd + r;
        atomicAdd(&lsum[h], sacc[r]);
        atomicAdd(&lsq[h],  sqacc[r]);
    }
    __syncthreads();
    if (t < HH) {
        atomicAdd(&gsum[t], lsum[t]);
        atomicAdd(&gsq[t],  lsq[t]);
    }
}

// ---------------------------------------------------------------------------
// Kernel 3: InstanceNorm (affine=False) + ReLU, in place over d_out
// ---------------------------------------------------------------------------
__global__ __launch_bounds__(256) void norm_relu_kernel(
    float* __restrict__ y, const float* __restrict__ gsum, const float* __restrict__ gsq)
{
    const int h = blockIdx.y;
    const float inv = 1.0f / 1048576.0f;           // 1/(N*N)
    const float mu  = gsum[h] * inv;
    const float var = gsq[h] * inv - mu * mu;
    const float sc  = rsqrtf(var + 1e-5f);

    const size_t base = ((size_t)h << 20) + ((size_t)blockIdx.x << 10) + ((size_t)threadIdx.x << 2);
    float4 v = *(float4*)(y + base);
    v.x = fmaxf(0.0f, (v.x - mu) * sc);
    v.y = fmaxf(0.0f, (v.y - mu) * sc);
    v.z = fmaxf(0.0f, (v.z - mu) * sc);
    v.w = fmaxf(0.0f, (v.w - mu) * sc);
    *(float4*)(y + base) = v;
}

extern "C" void kernel_launch(void* const* d_in, const int* in_sizes, int n_in,
                              void* d_out, int out_size, void* d_ws, size_t ws_size,
                              hipStream_t stream) {
    const float* z0 = (const float*)d_in[0];   // (1, 1024, 64)
    const float* W  = (const float*)d_in[1];   // (64, 128)
    // d_in[2] = bias b[h]: constant per channel -> cancels under InstanceNorm, unused.
    float* out = (float*)d_out;                // (1, 64, 1024, 1024)

    char*   ws   = (char*)d_ws;
    float*  z_t  = (float*)ws;                  // 64*1024 f32  = 256 KB
    __bf16* w_bf = (__bf16*)(ws + 262144);      // 64*128 bf16  =  16 KB
    float*  gsum = (float*)(ws + 278528);       // 64 f32
    float*  gsq  = (float*)(ws + 278784);       // 64 f32

    prep_kernel<<<256, 256, 0, stream>>>(z0, W, z_t, w_bf, gsum, gsq);

    dim3 g1(NN, NN / MB);                       // 1024 x 8 blocks
    pairwise_gemm_kernel<<<g1, 256, 0, stream>>>(z_t, w_bf, out, gsum, gsq);

    dim3 g2(1024, HH);
    norm_relu_kernel<<<g2, 256, 0, stream>>>(out, gsum, gsq);
}
